// _SplineKernelAttention_21071109554240
// MI455X (gfx1250) — matrix operand order repaired and checked by reading, not compiled, not run on hardware
//
#include <hip/hip_runtime.h>
#include <hip/hip_bf16.h>

// ---------------------------------------------------------------------------
// SplineKernelAttention for MI455X (gfx1250, wave32, WMMA)
//
// Pipeline:
//   1) cvt:    tok, Wv, Wout  f32 -> f16 (for WMMA GEMMs)
//   2) head_vec: aq[h,:] = wq1 . Wq[h*256:(h+1)*256, :]  (same for ak)  —
//      q/k never materialized (associativity of the rank-1 contraction)
//   3) proj_uv: uq[b,h,l] = tok[b,l,:] . aq[h,:]  (and uk)
//   4) gemm_wmma: v = tok @ Wv.T  (f16 in, f32 out, v_wmma_f32_16x16x32_f16)
//   5) topk_attn: spline scores over 4096 keys, exact top-32 via lane-local
//      LDS buffers + wave merge, softmax, coalesced v-gather -> ctx (f16)
//   6) gemm_wmma: out = ctx @ Wout.T  -> d_out (f32)
// ---------------------------------------------------------------------------

typedef __attribute__((ext_vector_type(16))) _Float16 v16h;
typedef __attribute__((ext_vector_type(8)))  float    v8f;

#define NEG_BIG (-3.402823466e38f)

union Frag {
    v16h h;
    uint4 q[2];
};

// ---------------------------------------------------------------- cvt f32->f16
__global__ __launch_bounds__(256) void f32_to_f16_kernel(const float* __restrict__ in,
                                                         _Float16* __restrict__ out, int n) {
    int i = blockIdx.x * 256 + threadIdx.x;
    if (i < n) out[i] = (_Float16)in[i];
}

// ------------------------------------------------- per-head projection vectors
// aq[h*768+k] = sum_j wq1[j] * Wq[(h*256+j)*768 + k]   (likewise ak from Wk)
__global__ __launch_bounds__(256) void head_vec_kernel(const float* __restrict__ Wq,
                                                       const float* __restrict__ Wk,
                                                       const float* __restrict__ wq1,
                                                       const float* __restrict__ wk1,
                                                       float* __restrict__ aq,
                                                       float* __restrict__ ak) {
    int i = blockIdx.x * 256 + threadIdx.x;          // 0 .. 2*3*768-1
    if (i >= 2 * 3 * 768) return;
    int which = i / (3 * 768);
    int rem   = i % (3 * 768);
    int h = rem / 768, kcol = rem % 768;
    const float* W  = which ? Wk  : Wq;
    const float* w1 = which ? wk1 : wq1;
    float acc = 0.0f;
    for (int j = 0; j < 256; ++j)
        acc += w1[j] * W[(size_t)(h * 256 + j) * 768 + kcol];
    (which ? ak : aq)[h * 768 + kcol] = acc;
}

// ----------------------------------------------------------------- uq/uk rows
// one wave per (b,l) row; lanes stride over k; shuffle reduce 6 accumulators
__global__ __launch_bounds__(256) void proj_uv_kernel(const float* __restrict__ tok,
                                                      const float* __restrict__ aq,
                                                      const float* __restrict__ ak,
                                                      float* __restrict__ uq,
                                                      float* __restrict__ uk) {
    int g    = blockIdx.x * 8 + (threadIdx.x >> 5);  // b*L + l, 0..8191
    int lane = threadIdx.x & 31;
    const float* row = tok + (size_t)g * 768;
    float sq0 = 0, sq1 = 0, sq2 = 0, sk0 = 0, sk1 = 0, sk2 = 0;
    for (int k = lane; k < 768; k += 32) {
        float t = row[k];
        sq0 += t * aq[k];        sq1 += t * aq[768 + k];  sq2 += t * aq[1536 + k];
        sk0 += t * ak[k];        sk1 += t * ak[768 + k];  sk2 += t * ak[1536 + k];
    }
    for (int off = 16; off; off >>= 1) {
        sq0 += __shfl_xor(sq0, off); sq1 += __shfl_xor(sq1, off); sq2 += __shfl_xor(sq2, off);
        sk0 += __shfl_xor(sk0, off); sk1 += __shfl_xor(sk1, off); sk2 += __shfl_xor(sk2, off);
    }
    if (lane == 0) {
        int b = g >> 12, l = g & 4095;
        uq[((size_t)(b * 3 + 0)) * 4096 + l] = sq0;
        uq[((size_t)(b * 3 + 1)) * 4096 + l] = sq1;
        uq[((size_t)(b * 3 + 2)) * 4096 + l] = sq2;
        uk[((size_t)(b * 3 + 0)) * 4096 + l] = sk0;
        uk[((size_t)(b * 3 + 1)) * 4096 + l] = sk1;
        uk[((size_t)(b * 3 + 2)) * 4096 + l] = sk2;
    }
}

// --------------------------------------------------------------- WMMA GEMM
// C[M,N] (f32) = A[M,K] (f16, row-major) @ W[N,K]^T (f16, row-major)
// block: 256 thr = 8 waves in 4x2; each wave: 32x32 tile = 2x2 wmma 16x16x32
// A frag (lane): A[m][k0..k0+7], A[m][k0+16..k0+23], k0 = (lane>=16)?8:0
// B frag (lane): W[n][k0..k0+7], W[n][k0+16..k0+23], k0 = (lane>=16)?8:0
__global__ __launch_bounds__(256) void gemm_wmma_kernel(const _Float16* __restrict__ A,
                                                        const _Float16* __restrict__ W,
                                                        float* __restrict__ C,
                                                        int M, int N, int K) {
    const int lane  = threadIdx.x & 31;
    const int wave  = threadIdx.x >> 5;
    const int wm    = wave & 3;         // 4 waves along M (4*32 = 128)
    const int wn    = wave >> 2;        // 2 waves along N (2*32 = 64)
    const int lhalf = lane >> 4;        // 0 | 1
    const int lmod  = lane & 15;
    const int rowBase = blockIdx.x * 128 + wm * 32;
    const int colBase = blockIdx.y * 64  + wn * 32;

    const _Float16* a0 = A + (size_t)(rowBase +      lmod) * K;
    const _Float16* a1 = A + (size_t)(rowBase + 16 + lmod) * K;
    const _Float16* b0 = W + (size_t)(colBase +      lmod) * K;
    const _Float16* b1 = W + (size_t)(colBase + 16 + lmod) * K;
    const int ka = lhalf * 8;
    const int kb = lhalf * 16;

    v8f acc00 = {}, acc01 = {}, acc10 = {}, acc11 = {};

    for (int kk = 0; kk < K; kk += 32) {
        Frag fa0, fa1, fb0, fb1;
        fa0.q[0] = *(const uint4*)(a0 + kk + ka);
        fa0.q[1] = *(const uint4*)(a0 + kk + ka + 16);
        fa1.q[0] = *(const uint4*)(a1 + kk + ka);
        fa1.q[1] = *(const uint4*)(a1 + kk + ka + 16);
        fb0.q[0] = *(const uint4*)(b0 + kk + ka);
        fb0.q[1] = *(const uint4*)(b0 + kk + ka + 16);
        fb1.q[0] = *(const uint4*)(b1 + kk + ka);
        fb1.q[1] = *(const uint4*)(b1 + kk + ka + 16);

        acc00 = __builtin_amdgcn_wmma_f32_16x16x32_f16(false, fa0.h, false, fb0.h,
                                                       (short)0, acc00, false, false);
        acc01 = __builtin_amdgcn_wmma_f32_16x16x32_f16(false, fa0.h, false, fb1.h,
                                                       (short)0, acc01, false, false);
        acc10 = __builtin_amdgcn_wmma_f32_16x16x32_f16(false, fa1.h, false, fb0.h,
                                                       (short)0, acc10, false, false);
        acc11 = __builtin_amdgcn_wmma_f32_16x16x32_f16(false, fa1.h, false, fb1.h,
                                                       (short)0, acc11, false, false);
    }

    // C/D layout: VGPR r -> row r (lanes 0-15) / row r+8 (lanes 16-31), col = lane&15
    const int r0 = lhalf * 8;
    {
        int row = rowBase + r0, col = colBase + lmod;
#pragma unroll
        for (int r = 0; r < 8; ++r) C[(size_t)(row + r) * N + col] = acc00[r];
    }
    {
        int row = rowBase + r0, col = colBase + 16 + lmod;
#pragma unroll
        for (int r = 0; r < 8; ++r) C[(size_t)(row + r) * N + col] = acc01[r];
    }
    {
        int row = rowBase + 16 + r0, col = colBase + lmod;
#pragma unroll
        for (int r = 0; r < 8; ++r) C[(size_t)(row + r) * N + col] = acc10[r];
    }
    {
        int row = rowBase + 16 + r0, col = colBase + 16 + lmod;
#pragma unroll
        for (int r = 0; r < 8; ++r) C[(size_t)(row + r) * N + col] = acc11[r];
    }
}

// ------------------------------------------------------- spline top-k + gather
// one wave per query (b,h,l); 8 waves / block.
// Phase 1: lane-local exact top-32 over a 128-key strided subset (LDS slots)
// Phase 2: 32-round wave argmax merge -> lane j holds rank-j (val, idx)
// Phase 3: wave softmax; Phase 4: coalesced gather of v rows -> ctx (f16)
__global__ __launch_bounds__(256) void topk_attn_kernel(const float* __restrict__ uq,
                                                        const float* __restrict__ uk,
                                                        const float* __restrict__ tau_row,
                                                        const float* __restrict__ coeff,
                                                        const float* __restrict__ v,
                                                        _Float16* __restrict__ ctx) {
    __shared__ float          cv[256 * 32];   // candidate values  (32 KB)
    __shared__ unsigned short ci[256 * 32];   // candidate keys    (16 KB)
    __shared__ float attnS[8][32];
    __shared__ int   idxS[8][32];

    const int lane = threadIdx.x & 31;
    const int wave = threadIdx.x >> 5;
    const int g = blockIdx.x * 8 + wave;       // b*H*L + h*L + l
    const int l = g & 4095;
    const int h = (g >> 12) % 3;
    const int b = g / (3 * 4096);

    const float* ukrow = uk + (size_t)(b * 3 + h) * 4096;
    const float  uqv    = uq[(size_t)(b * 3 + h) * 4096 + l];
    const float  invtau = 1.0f / (tau_row[b * 4096 + l] + 1e-6f);

    float cf[8];
#pragma unroll
    for (int n = 0; n < 8; ++n) cf[n] = coeff[h * 8 + n];
    const float delta  = 6.0f / 7.0f;
    const float invden = 1.0f / (delta + 1e-6f);

    const int base = threadIdx.x * 32;
#pragma unroll
    for (int i = 0; i < 32; ++i) cv[base + i] = NEG_BIG;

    // ---- phase 1: lane-local top-32 (exact; every lane sees 128 keys)
    float cmin = NEG_BIG;
    int   cpos = 0;
    for (int t = 0; t < 128; ++t) {
        int key = (t << 5) | lane;
        float diff = uqv - ukrow[key];
        float s = 0.0f;
#pragma unroll
        for (int n = 0; n < 8; ++n) {
            float w = 1.0f - fabsf(diff - (-3.0f + (float)n * delta)) * invden;
            s += cf[n] * fmaxf(w, 0.0f);
        }
        s *= invtau;
        if (s > cmin) {
            cv[base + cpos] = s;
            ci[base + cpos] = (unsigned short)key;
            float m = cv[base]; int mp = 0;
            for (int i = 1; i < 32; ++i) {
                float x = cv[base + i];
                if (x < m) { m = x; mp = i; }
            }
            cmin = m; cpos = mp;
        }
    }
    __syncthreads();

    // ---- phase 2: select global top-32 of this wave's 1024 candidates
    const int waveBase = wave * 32 * 32;
    float selv = 0.0f;
    int   seli = 0;
    for (int j = 0; j < 32; ++j) {
        float bv = cv[base]; int bp = 0;
        for (int i = 1; i < 32; ++i) {
            float x = cv[base + i];
            if (x > bv) { bv = x; bp = i; }
        }
        int bl = (lane << 5) | bp;
        for (int off = 16; off; off >>= 1) {
            float ov = __shfl_xor(bv, off);
            int   ol = __shfl_xor(bl, off);
            if (ov > bv) { bv = ov; bl = ol; }
        }
        int key = ci[waveBase + bl];           // broadcast read
        if (lane == j) { selv = bv; seli = key; }
        if (lane == (bl >> 5)) cv[waveBase + bl] = NEG_BIG;  // knock out winner
    }

    // ---- phase 3: softmax across the 32 selected scores (lane j = rank j)
    float mx = selv;
    for (int off = 16; off; off >>= 1) mx = fmaxf(mx, __shfl_xor(mx, off));
    float e = __expf(selv - mx);
    float sm = e;
    for (int off = 16; off; off >>= 1) sm += __shfl_xor(sm, off);
    attnS[wave][lane] = e / sm;
    idxS[wave][lane]  = seli;
    __syncthreads();

    // ---- phase 4: ctx[b,l,h*256+dd] = sum_j attn_j * v[b, idx_j, h*256+dd]
    float acc[8] = {0, 0, 0, 0, 0, 0, 0, 0};
    const float* vb = v + (size_t)(b * 4096) * 768 + h * 256;
    for (int j = 0; j < 32; ++j) {
        float a = attnS[wave][j];
        const float* vr = vb + (size_t)idxS[wave][j] * 768;
#pragma unroll
        for (int r = 0; r < 8; ++r) acc[r] += a * vr[lane + r * 32];  // 128B coalesced
    }
    _Float16* crow = ctx + (size_t)(b * 4096 + l) * 768 + h * 256;
#pragma unroll
    for (int r = 0; r < 8; ++r) crow[lane + r * 32] = (_Float16)acc[r];
}

// ---------------------------------------------------------------------------
extern "C" void kernel_launch(void* const* d_in, const int* in_sizes, int n_in,
                              void* d_out, int out_size, void* d_ws, size_t ws_size,
                              hipStream_t stream) {
    const float* tok   = (const float*)d_in[0];
    const float* tau   = (const float*)d_in[1];
    const float* Wq    = (const float*)d_in[2];
    const float* Wk    = (const float*)d_in[3];
    const float* Wv    = (const float*)d_in[4];
    const float* wq1   = (const float*)d_in[5];
    const float* wk1   = (const float*)d_in[6];
    const float* coeff = (const float*)d_in[7];
    const float* Wout  = (const float*)d_in[8];

    const int M = 2 * 4096;   // B*L = 8192
    const int D = 768;

    char*  ws  = (char*)d_ws;
    size_t off = 0;
    auto alloc = [&](size_t bytes) -> void* {
        void* p = ws + off;
        off = (off + bytes + 255) & ~(size_t)255;
        return p;
    };

    _Float16* tokh  = (_Float16*)alloc((size_t)M * D * 2);
    _Float16* Wvh   = (_Float16*)alloc((size_t)D * D * 2);
    _Float16* Wouth = (_Float16*)alloc((size_t)D * D * 2);
    float*    aq    = (float*)alloc((size_t)3 * D * 4);
    float*    ak    = (float*)alloc((size_t)3 * D * 4);
    float*    uqb   = (float*)alloc((size_t)2 * 3 * 4096 * 4);
    float*    ukb   = (float*)alloc((size_t)2 * 3 * 4096 * 4);
    float*    vbuf  = (float*)alloc((size_t)M * D * 4);
    _Float16* ctxh  = (_Float16*)alloc((size_t)M * D * 2);

    // 1) convert tok / Wv / Wout to f16
    {
        int n = M * D;
        f32_to_f16_kernel<<<(n + 255) / 256, 256, 0, stream>>>(tok, tokh, n);
        n = D * D;
        f32_to_f16_kernel<<<(n + 255) / 256, 256, 0, stream>>>(Wv, Wvh, n);
        f32_to_f16_kernel<<<(n + 255) / 256, 256, 0, stream>>>(Wout, Wouth, n);
    }
    // 2) per-head projection vectors (q,k fused away)
    head_vec_kernel<<<18, 256, 0, stream>>>(Wq, Wk, wq1, wk1, aq, ak);
    // 3) uq / uk
    proj_uv_kernel<<<M / 8, 256, 0, stream>>>(tok, aq, ak, uqb, ukb);
    // 4) v = tok @ Wv.T  (WMMA)
    gemm_wmma_kernel<<<dim3(M / 128, D / 64), 256, 0, stream>>>(tokh, Wvh, vbuf, M, D, D);
    // 5) spline top-k attention -> ctx (f16)
    topk_attn_kernel<<<(2 * 3 * 4096) / 8, 256, 0, stream>>>(uqb, ukb, tau, coeff, vbuf, ctxh);
    // 6) out = ctx @ Wout.T  (WMMA)
    gemm_wmma_kernel<<<dim3(M / 128, D / 64), 256, 0, stream>>>(ctxh, Wouth, (float*)d_out, M, D, D);
}
